// AdvancedGraphResBlock_7224134992065
// MI455X (gfx1250) — compile-verified, matrix operand order checked by reading, not run on hardware
//
#include <hip/hip_runtime.h>
#include <hip/hip_bf16.h>

typedef _Float16 h16;
typedef __attribute__((ext_vector_type(16))) _Float16 v16h;
typedef __attribute__((ext_vector_type(8)))  float    v8f;

// Problem constants
constexpr int Bb = 4, Nn = 4096, Dd = 128, Tt = 128, Hh = 4, HDc = 32;
constexpr int BN = Bb * Nn;           // 16384 tokens
constexpr float NEGBIG = -1e9f;

// ---------------------------------------------------------------------------
// WMMA helpers (wave32, 16x16x32 f16 -> f32)
// ---------------------------------------------------------------------------
__device__ __forceinline__ v8f wmma16(v16h a, v16h b, v8f c) {
  return __builtin_amdgcn_wmma_f32_16x16x32_f16(false, a, false, b, (short)0, c,
                                                false, false);
}

// A: row-major [16][128] tile base (lda=128, f16), BT: [16 cols][128] (WT rows)
__device__ __forceinline__ v8f gemm_k128(const h16* A, const h16* BT, int lane,
                                         v8f acc) {
  const int lm = lane & 15, lh = lane >> 4;
#pragma unroll
  for (int k0 = 0; k0 < 128; k0 += 32) {
    v16h a = *(const v16h*)(A + lm * 128 + lh * 16 + k0);
    v16h b = *(const v16h*)(BT + lm * 128 + lh * 16 + k0);
    acc = wmma16(a, b, acc);
  }
  return acc;
}

// ---------------------------------------------------------------------------
// Prep: weight transpose to f16, adjacency bit-packing
// ---------------------------------------------------------------------------
__global__ void wT_kernel(const float* __restrict__ W, h16* __restrict__ WT,
                          int K, int Nc) {
  int idx = blockIdx.x * blockDim.x + threadIdx.x;
  if (idx >= K * Nc) return;
  int n = idx / K, k = idx % K;     // WT[n*K + k] = W[k*Nc + n]
  WT[idx] = (h16)W[k * Nc + n];
}

__global__ void adjbits_kernel(const int* __restrict__ adj,
                               unsigned* __restrict__ bits) {
  int idx = blockIdx.x * blockDim.x + threadIdx.x;
  if (idx >= Nn * (Nn / 32)) return;
  int i = idx >> 7, w = idx & 127;
  const int* row = adj + (size_t)i * Nn + w * 32;
  unsigned v = 0;
#pragma unroll
  for (int t = 0; t < 32; ++t) v |= (row[t] != 0 ? 1u : 0u) << t;
  bits[idx] = v;
}

// ---------------------------------------------------------------------------
// Time MLP: t_params = mish(t_emb) @ Wt + bt   [B, 2D]
// ---------------------------------------------------------------------------
__global__ void time_mlp_kernel(const float* __restrict__ t_emb,
                                const float* __restrict__ Wt,
                                const float* __restrict__ bt,
                                float* __restrict__ tp) {
  __shared__ float m[Tt];
  int b = blockIdx.x, j = threadIdx.x;  // 256 threads
  if (j < Tt) {
    float x = t_emb[b * Tt + j];
    float sp = (x > 20.f) ? x : log1pf(expf(x));
    m[j] = x * tanhf(sp);
  }
  __syncthreads();
  float acc = bt[j];
  for (int t = 0; t < Tt; ++t) acc += m[t] * Wt[t * 256 + j];
  tp[b * 256 + j] = acc;
}

// ---------------------------------------------------------------------------
// FiLM + LayerNorm -> f16 (one wave per row). tp==null -> plain LN.
// ---------------------------------------------------------------------------
__global__ void ln_kernel(const float* __restrict__ x,
                          const float* __restrict__ tp,
                          const float* __restrict__ g,
                          const float* __restrict__ be,
                          h16* __restrict__ out) {
  int wid = threadIdx.x >> 5, lane = threadIdx.x & 31;
  int row = blockIdx.x * 8 + wid;
  if (row >= BN) return;
  int b = row >> 12;
  const float* xr = x + (size_t)row * Dd;
  float v[4];
  float s = 0.f;
#pragma unroll
  for (int i = 0; i < 4; ++i) {
    int d = i * 32 + lane;
    float r = xr[d];
    if (tp) r = r * (1.f + tp[b * 256 + d]) + tp[b * 256 + 128 + d];
    v[i] = r;
    s += r;
  }
#pragma unroll
  for (int off = 16; off; off >>= 1) s += __shfl_xor(s, off, 32);
  float mu = s * (1.f / 128.f);
  float ss = 0.f;
#pragma unroll
  for (int i = 0; i < 4; ++i) {
    float d = v[i] - mu;
    ss += d * d;
  }
#pragma unroll
  for (int off = 16; off; off >>= 1) ss += __shfl_xor(ss, off, 32);
  float rstd = rsqrtf(ss * (1.f / 128.f) + 1e-5f);
#pragma unroll
  for (int i = 0; i < 4; ++i) {
    int d = i * 32 + lane;
    out[(size_t)row * Dd + d] = (h16)((v[i] - mu) * rstd * g[d] + be[d]);
  }
}

// ---------------------------------------------------------------------------
// GEMM kernels: each block = one 16-row tile, 8 waves = 8 col tiles of 16
// ---------------------------------------------------------------------------
__global__ void gemm_h1_kernel(const h16* __restrict__ h0,
                               const h16* __restrict__ W1T,
                               const float* __restrict__ b1,
                               h16* __restrict__ h1) {
  int mt = blockIdx.x, w = threadIdx.x >> 5, lane = threadIdx.x & 31;
  v8f acc = {};
  acc = gemm_k128(h0 + (size_t)mt * 16 * 128, W1T + w * 16 * 128, lane, acc);
  int n = w * 16 + (lane & 15);
  float bias = b1[n];
  int mbase = mt * 16 + 8 * (lane >> 4);
#pragma unroll
  for (int r = 0; r < 8; ++r)
    h1[(size_t)(mbase + r) * 128 + n] = (h16)(acc[r] + bias);
}

__global__ void gemm_glu_kernel(const h16* __restrict__ h1,
                                const h16* __restrict__ WgT,
                                const float* __restrict__ bg,
                                h16* __restrict__ h2) {
  int mt = blockIdx.x, w = threadIdx.x >> 5, lane = threadIdx.x & 31;
  const int lm = lane & 15, lh = lane >> 4;
  const h16* A = h1 + (size_t)mt * 16 * 128;
  const h16* Ba = WgT + w * 16 * 128;          // cols [w*16, w*16+16)
  const h16* Bg = WgT + (w + 8) * 16 * 128;    // cols [128+w*16, ...)
  v8f aa = {}, gg = {};
#pragma unroll
  for (int k0 = 0; k0 < 128; k0 += 32) {
    v16h a = *(const v16h*)(A + lm * 128 + lh * 16 + k0);
    v16h b0 = *(const v16h*)(Ba + lm * 128 + lh * 16 + k0);
    v16h b1 = *(const v16h*)(Bg + lm * 128 + lh * 16 + k0);
    aa = wmma16(a, b0, aa);
    gg = wmma16(a, b1, gg);
  }
  int n = w * 16 + lm;
  float ba = bg[n], bgv = bg[128 + n];
  int mbase = mt * 16 + 8 * lh;
#pragma unroll
  for (int r = 0; r < 8; ++r) {
    float av = aa[r] + ba;
    float gv = gg[r] + bgv;
    h2[(size_t)(mbase + r) * 128 + n] = (h16)(av * (1.f / (1.f + expf(-gv))));
  }
}

__global__ void gemm_xmid_kernel(const h16* __restrict__ h2,
                                 const h16* __restrict__ W2T,
                                 const float* __restrict__ b2,
                                 const float* __restrict__ x,
                                 float* __restrict__ xmid) {
  int mt = blockIdx.x, w = threadIdx.x >> 5, lane = threadIdx.x & 31;
  v8f acc = {};
  acc = gemm_k128(h2 + (size_t)mt * 16 * 128, W2T + w * 16 * 128, lane, acc);
  int n = w * 16 + (lane & 15);
  float bias = b2[n];
  int mbase = mt * 16 + 8 * (lane >> 4);
#pragma unroll
  for (int r = 0; r < 8; ++r) {
    size_t idx = (size_t)(mbase + r) * 128 + n;
    xmid[idx] = x[idx] + acc[r] + bias;
  }
}

// blockIdx.y: 0=Q, 1=K, 2=V(transposed store)
__global__ void gemm_qkv_kernel(const h16* __restrict__ xn,
                                const h16* __restrict__ WqT,
                                const h16* __restrict__ WkT,
                                const h16* __restrict__ WvT,
                                const float* __restrict__ bq,
                                const float* __restrict__ bk,
                                const float* __restrict__ bv,
                                h16* __restrict__ qb, h16* __restrict__ kb,
                                h16* __restrict__ vT) {
  int mt = blockIdx.x, which = blockIdx.y;
  int w = threadIdx.x >> 5, lane = threadIdx.x & 31;
  const h16* WT = which == 0 ? WqT : (which == 1 ? WkT : WvT);
  const float* bias = which == 0 ? bq : (which == 1 ? bk : bv);
  v8f acc = {};
  acc = gemm_k128(xn + (size_t)mt * 16 * 128, WT + w * 16 * 128, lane, acc);
  int n = w * 16 + (lane & 15);
  float bv_ = bias[n];
  int mbase = mt * 16 + 8 * (lane >> 4);
  if (which < 2) {
    h16* dst = which == 0 ? qb : kb;
#pragma unroll
    for (int r = 0; r < 8; ++r)
      dst[(size_t)(mbase + r) * 128 + n] = (h16)(acc[r] + bv_);
  } else {
#pragma unroll
    for (int r = 0; r < 8; ++r) {
      int m = mbase + r;
      int b = m >> 12, tok = m & (Nn - 1);
      // vT layout: [B][H][HD][N]
      size_t idx = ((size_t)((b * Hh + (n >> 5)) * HDc + (n & 31))) * Nn + tok;
      vT[idx] = (h16)(acc[r] + bv_);
    }
  }
}

// ---------------------------------------------------------------------------
// Flash attention with adjacency bitmask.
// One wave = (b, h, 32-query block = 2 tiles of 16). Per 32-key step:
//   2x2 score WMMAs, 2x2 PV WMMAs, 2x1 row-sum WMMAs (P @ ones),
//   1 adjacency word per row, shuffles only for the running max.
// ---------------------------------------------------------------------------
__global__ void attn_kernel(const h16* __restrict__ q,
                            const h16* __restrict__ kk,
                            const h16* __restrict__ vT,
                            const unsigned* __restrict__ adjb,
                            h16* __restrict__ ao) {
  __shared__ __align__(128) h16 ldsP[8][2][16 * 32];
  int wid = threadIdx.x >> 5, lane = threadIdx.x & 31;
  int task = blockIdx.x * 8 + wid;          // B*H*(N/32) = 2048 tasks
  int qt2 = task & 127;                     // 32-query block
  int bh = task >> 7;
  int b = bh >> 2, h = bh & 3;
  int qbase = qt2 * 32;
  const int lm = lane & 15, lh = lane >> 4;
  const float sc = 0.17677669529663687f;    // HD^-0.5

  // ones B-fragment: P @ ones(32x16) -> every column = row-sum of P
  v16h ones;
#pragma unroll
  for (int i = 0; i < 16; ++i) ones[i] = (h16)1.0f;

  // Q fragments (loop invariant)
  v16h Qf[2];
#pragma unroll
  for (int t = 0; t < 2; ++t)
    Qf[t] = *(const v16h*)(q + ((size_t)(b * Nn + qbase + t * 16 + lm)) * 128 +
                           h * 32 + lh * 16);

  v8f o0[2] = {}, o1[2] = {}, lacc[2] = {};
  float mrow[2][8];
#pragma unroll
  for (int t = 0; t < 2; ++t)
#pragma unroll
    for (int r = 0; r < 8; ++r) mrow[t][r] = -3.0e38f;

  const h16* vbase = vT + ((size_t)((b * Hh + h) * HDc)) * Nn;

  for (int kt = 0; kt < Nn / 32; ++kt) {
    int kbse = kt * 32;
    // --- shared K fragments for both query tiles ---
    v16h K0 = *(const v16h*)(kk + ((size_t)(b * Nn + kbse + lm)) * 128 +
                             h * 32 + lh * 16);
    v16h K1 = *(const v16h*)(kk + ((size_t)(b * Nn + kbse + 16 + lm)) * 128 +
                             h * 32 + lh * 16);
#pragma unroll
    for (int t = 0; t < 2; ++t) {
      v8f s0 = {}, s1 = {};
      s0 = wmma16(Qf[t], K0, s0);
      s1 = wmma16(Qf[t], K1, s1);
      // mask + scale: one adjacency word per row covers all 32 keys
      float pv0[8], pv1[8], tmax[8];
#pragma unroll
      for (int r = 0; r < 8; ++r) {
        int i = qbase + t * 16 + r + 8 * lh;
        unsigned word = adjb[(size_t)i * 128 + kt];
        float v0 = ((word >> lm) & 1u) ? s0[r] * sc : NEGBIG;
        float v1 = ((word >> (16 + lm)) & 1u) ? s1[r] * sc : NEGBIG;
        pv0[r] = v0; pv1[r] = v1;
        tmax[r] = fmaxf(v0, v1);
      }
      // row max over the 16 lanes of each half-wave
#pragma unroll
      for (int r = 0; r < 8; ++r) {
#pragma unroll
        for (int off = 8; off; off >>= 1)
          tmax[r] = fmaxf(tmax[r], __shfl_xor(tmax[r], off, 16));
      }
      // online rescale (sum handled by WMMA ones-column below)
      h16* lp = ldsP[wid][t];
#pragma unroll
      for (int r = 0; r < 8; ++r) {
        float mn = fmaxf(mrow[t][r], tmax[r]);
        float cf = __expf(mrow[t][r] - mn);
        mrow[t][r] = mn;
        float p0 = __expf(pv0[r] - mn);
        float p1 = __expf(pv1[r] - mn);
        o0[t][r] *= cf;
        o1[t][r] *= cf;
        lacc[t][r] *= cf;
        int m = r + 8 * lh;
        lp[m * 32 + lm] = (h16)p0;
        lp[m * 32 + 16 + lm] = (h16)p1;
      }
    }
    asm volatile("s_wait_dscnt 0" ::: "memory");  // cross-lane LDS visibility
    // --- shared V fragments ---
    v16h V0 = *(const v16h*)(vbase + (size_t)lm * Nn + kbse + lh * 16);
    v16h V1 = *(const v16h*)(vbase + (size_t)(16 + lm) * Nn + kbse + lh * 16);
#pragma unroll
    for (int t = 0; t < 2; ++t) {
      v16h Pf = *(const v16h*)(ldsP[wid][t] + lm * 32 + lh * 16);
      o0[t] = wmma16(Pf, V0, o0[t]);
      o1[t] = wmma16(Pf, V1, o1[t]);
      lacc[t] = wmma16(Pf, ones, lacc[t]);   // row sums, rescaled like O
    }
  }
  // epilogue: normalize and store f16 [BN, D]
#pragma unroll
  for (int t = 0; t < 2; ++t) {
#pragma unroll
    for (int r = 0; r < 8; ++r) {
      float inv = 1.f / lacc[t][r];
      size_t m = (size_t)(b * Nn + qbase + t * 16 + r + 8 * lh);
      ao[m * 128 + h * 32 + lm] = (h16)(o0[t][r] * inv);
      ao[m * 128 + h * 32 + 16 + lm] = (h16)(o1[t][r] * inv);
    }
  }
}

// ---------------------------------------------------------------------------
// Output projection + residual
// ---------------------------------------------------------------------------
__global__ void gemm_out_kernel(const h16* __restrict__ ao,
                                const h16* __restrict__ WoT,
                                const float* __restrict__ bo,
                                const float* __restrict__ xmid,
                                float* __restrict__ out) {
  int mt = blockIdx.x, w = threadIdx.x >> 5, lane = threadIdx.x & 31;
  v8f acc = {};
  acc = gemm_k128(ao + (size_t)mt * 16 * 128, WoT + w * 16 * 128, lane, acc);
  int n = w * 16 + (lane & 15);
  float bias = bo[n];
  int mbase = mt * 16 + 8 * (lane >> 4);
#pragma unroll
  for (int r = 0; r < 8; ++r) {
    size_t idx = (size_t)(mbase + r) * 128 + n;
    out[idx] = xmid[idx] + acc[r] + bias;
  }
}

// ---------------------------------------------------------------------------
// Host launcher
// ---------------------------------------------------------------------------
extern "C" void kernel_launch(void* const* d_in, const int* in_sizes, int n_in,
                              void* d_out, int out_size, void* d_ws,
                              size_t ws_size, hipStream_t stream) {
  const float* x = (const float*)d_in[0];
  const float* t_emb = (const float*)d_in[1];
  const int* adj = (const int*)d_in[2];
  const float* Wt = (const float*)d_in[3];
  const float* bt = (const float*)d_in[4];
  const float* W1 = (const float*)d_in[5];
  const float* b1 = (const float*)d_in[6];
  const float* Wg = (const float*)d_in[7];
  const float* bg = (const float*)d_in[8];
  const float* W2 = (const float*)d_in[9];
  const float* b2 = (const float*)d_in[10];
  const float* Wq = (const float*)d_in[11];
  const float* bq = (const float*)d_in[12];
  const float* Wk = (const float*)d_in[13];
  const float* bk = (const float*)d_in[14];
  const float* Wv = (const float*)d_in[15];
  const float* bv = (const float*)d_in[16];
  const float* Wo = (const float*)d_in[17];
  const float* bo = (const float*)d_in[18];
  const float* g1 = (const float*)d_in[19];
  const float* be1 = (const float*)d_in[20];
  const float* g2 = (const float*)d_in[21];
  const float* be2 = (const float*)d_in[22];
  float* out = (float*)d_out;

  // workspace carving (256B aligned)
  char* p = (char*)d_ws;
  auto carve = [&](size_t bytes) {
    void* r = (void*)p;
    p += (bytes + 255) & ~(size_t)255;
    return r;
  };
  h16* W1T = (h16*)carve(128 * 128 * sizeof(h16));
  h16* WgT = (h16*)carve(256 * 128 * sizeof(h16));
  h16* W2T = (h16*)carve(128 * 128 * sizeof(h16));
  h16* WqT = (h16*)carve(128 * 128 * sizeof(h16));
  h16* WkT = (h16*)carve(128 * 128 * sizeof(h16));
  h16* WvT = (h16*)carve(128 * 128 * sizeof(h16));
  h16* WoT = (h16*)carve(128 * 128 * sizeof(h16));
  float* tp = (float*)carve(Bb * 256 * sizeof(float));
  unsigned* adjb = (unsigned*)carve((size_t)Nn * 128 * sizeof(unsigned));
  h16* h0 = (h16*)carve((size_t)BN * 128 * sizeof(h16));
  h16* h1 = (h16*)carve((size_t)BN * 128 * sizeof(h16));
  h16* h2 = (h16*)carve((size_t)BN * 128 * sizeof(h16));
  float* xmid = (float*)carve((size_t)BN * 128 * sizeof(float));
  h16* xn = (h16*)carve((size_t)BN * 128 * sizeof(h16));
  h16* qb = (h16*)carve((size_t)BN * 128 * sizeof(h16));
  h16* kb = (h16*)carve((size_t)BN * 128 * sizeof(h16));
  h16* vT = (h16*)carve((size_t)BN * 128 * sizeof(h16));
  h16* ao = (h16*)carve((size_t)BN * 128 * sizeof(h16));

  // prep
  wT_kernel<<<64, 256, 0, stream>>>(W1, W1T, 128, 128);
  wT_kernel<<<128, 256, 0, stream>>>(Wg, WgT, 128, 256);
  wT_kernel<<<64, 256, 0, stream>>>(W2, W2T, 128, 128);
  wT_kernel<<<64, 256, 0, stream>>>(Wq, WqT, 128, 128);
  wT_kernel<<<64, 256, 0, stream>>>(Wk, WkT, 128, 128);
  wT_kernel<<<64, 256, 0, stream>>>(Wv, WvT, 128, 128);
  wT_kernel<<<64, 256, 0, stream>>>(Wo, WoT, 128, 128);
  adjbits_kernel<<<(Nn * 128) / 256, 256, 0, stream>>>(adj, adjb);
  time_mlp_kernel<<<Bb, 256, 0, stream>>>(t_emb, Wt, bt, tp);

  // FiLM + LN1 -> h0 (f16)
  ln_kernel<<<BN / 8, 256, 0, stream>>>(x, tp, g1, be1, h0);
  // temporal MLP
  gemm_h1_kernel<<<BN / 16, 256, 0, stream>>>(h0, W1T, b1, h1);
  gemm_glu_kernel<<<BN / 16, 256, 0, stream>>>(h1, WgT, bg, h2);
  gemm_xmid_kernel<<<BN / 16, 256, 0, stream>>>(h2, W2T, b2, x, xmid);
  // LN2 -> xn
  ln_kernel<<<BN / 8, 256, 0, stream>>>(xmid, nullptr, g2, be2, xn);
  // QKV projections
  gemm_qkv_kernel<<<dim3(BN / 16, 3), 256, 0, stream>>>(
      xn, WqT, WkT, WvT, bq, bk, bv, qb, kb, vT);
  // flash attention with adjacency mask (2 query tiles per wave)
  attn_kernel<<<(Bb * Hh * (Nn / 32)) / 8, 256, 0, stream>>>(qb, kb, vT, adjb,
                                                             ao);
  // output projection + residual
  gemm_out_kernel<<<BN / 16, 256, 0, stream>>>(ao, WoT, bo, xmid, out);
}
